// Autoencoder_49993419325660
// MI455X (gfx1250) — matrix operand order repaired and checked by reading, not compiled, not run on hardware
//
#include <hip/hip_runtime.h>
#include <hip/hip_bf16.h>
#include <math.h>

#define BATCH 16384
#define DIN   2048

typedef __attribute__((ext_vector_type(16))) __bf16 v16bf;
typedef __attribute__((ext_vector_type(8)))  float  v8f;
typedef int v4i __attribute__((vector_size(16)));

union Frag {
    v16bf v;
    unsigned int u[8];
    unsigned short h[16];
};

static __device__ __forceinline__ unsigned short f2bf(float f) {
    unsigned int u = __float_as_uint(f);
    unsigned int r = (u + 0x7FFFu + ((u >> 16) & 1u)) >> 16;   // RNE
    return (unsigned short)r;
}
static __device__ __forceinline__ float bf2f(unsigned short h) {
    return __uint_as_float(((unsigned int)h) << 16);
}
static __device__ __forceinline__ float sigmoidf(float x) {
    return 1.0f / (1.0f + expf(-x));
}

// ---------------------------------------------------------------------------
// Async global->LDS DMA (gfx1250 GLOBAL_LOAD_ASYNC_TO_LDS_B128, ASYNCcnt).
// Builtin signature is (v4i as1*, v4i as3*, imm offset, imm cpol).
// Falls back to VGPR-staged copies if the builtins are unavailable.
// ---------------------------------------------------------------------------
#if __has_builtin(__builtin_amdgcn_global_load_async_to_lds_b128) && \
    __has_builtin(__builtin_amdgcn_s_wait_asynccnt)
#define HAVE_ASYNC_LDS 1
static __device__ __forceinline__ void async_copy_b128(const void* g, void* l) {
    __builtin_amdgcn_global_load_async_to_lds_b128(
        (__attribute__((address_space(1))) v4i*)g,
        (__attribute__((address_space(3))) v4i*)l, 0, 0);
}
static __device__ __forceinline__ void async_wait0() {
    __builtin_amdgcn_s_wait_asynccnt(0);
}
#else
#define HAVE_ASYNC_LDS 0
#endif

// Copy `units16` chunks of 16 bytes from contiguous global to contiguous LDS,
// cooperatively across 256 threads.
static __device__ __forceinline__ void tile_copy(const unsigned short* g, unsigned short* l,
                                                 int units16, int tid) {
#if HAVE_ASYNC_LDS
    for (int i = tid; i < units16; i += 256)
        async_copy_b128((const char*)g + (size_t)i * 16, (char*)l + (size_t)i * 16);
#else
    const unsigned int* gu = (const unsigned int*)g;
    unsigned int* lu = (unsigned int*)l;
    for (int i = tid; i < units16 * 4; i += 256) lu[i] = gu[i];
#endif
}
static __device__ __forceinline__ void tile_copy_fence() {
#if HAVE_ASYNC_LDS
    async_wait0();
#endif
    __syncthreads();
}

// A-fragment (16x32 bf16) from LDS tile laid out [row][k], stride = K (even).
// ISA 7.12.2: lanes 0-15 hold M=lane, k = (j>=4)*16 + (j&3)*2; lanes 16-31: +8.
static __device__ __forceinline__ void load_a_frag(Frag& f, const unsigned short* lds,
                                                   int row, int stride, int kk, int lane) {
    const int sel = lane >> 4;
#pragma unroll
    for (int j = 0; j < 8; ++j) {
        int k0 = ((j >> 2) << 4) + (sel << 3) + ((j & 3) << 1);
        f.u[j] = *(const unsigned int*)&lds[row * stride + kk + k0];
    }
}

// B-fragment (32x16 bf16) from LDS tile laid out [n][k], stride = K (even).
// B layout matches A: VGPR j holds K = (j>=4)*16 + (j&3)*2 (+1) for lanes 0-15; lanes 16-31: +8. N=lane&15.
static __device__ __forceinline__ void load_b_frag(Frag& f, const unsigned short* lds,
                                                   int n, int K, int kk, int lane) {
    const int sel = lane >> 4;
#pragma unroll
    for (int j = 0; j < 8; ++j) {
        int k0 = ((j >> 2) << 4) + (sel << 3) + ((j & 3) << 1);
        f.u[j] = *(const unsigned int*)&lds[n * K + kk + k0];
    }
}

static __device__ __forceinline__ v8f wmma_bf16(const Frag& a, const Frag& b, v8f c) {
    return __builtin_amdgcn_wmma_f32_16x16x32_bf16(false, a.v, false, b.v, (short)0, c, false, false);
}

static __device__ __forceinline__ void block_reduce_atomic(float v, float* red, int tid, float* target) {
    red[tid] = v;
    __syncthreads();
    for (int s = 128; s > 0; s >>= 1) {
        if (tid < s) red[tid] += red[tid + s];
        __syncthreads();
    }
    if (tid == 0) atomicAdd(target, red[0]);
    __syncthreads();
}

// ---------------------------------------------------------------------------
// Weight packing: f32 -> bf16. Native [out][in] row-major layouts are exactly
// the [n][k] layout our B-fragments consume for the X @ W.T GEMMs.
// ---------------------------------------------------------------------------
__global__ void pack_kernel(const float* __restrict__ eW1, const float* __restrict__ eW2,
                            const float* __restrict__ eW3, const float* __restrict__ dW2,
                            const float* __restrict__ dW3, const float* __restrict__ dW4,
                            unsigned short* W1bf, unsigned short* W2bf, unsigned short* W3bf,
                            unsigned short* dW2bf, unsigned short* dW3bf, unsigned short* dW4bf,
                            float* accum) {
    const int stride = gridDim.x * blockDim.x;
    const int i0 = blockIdx.x * blockDim.x + threadIdx.x;
    if (i0 < 16) accum[i0] = 0.0f;
    for (int i = i0; i < 128 * 2048; i += stride) {          // drop treatment column -> stride 2048
        int n = i >> 11, k = i & 2047;
        W1bf[i] = f2bf(eW1[(size_t)n * 2049 + k]);
    }
    for (int i = i0; i < 64 * 128; i += stride)  W2bf[i]  = f2bf(eW2[i]);
    for (int i = i0; i < 32 * 64;  i += stride)  W3bf[i]  = f2bf(eW3[i]);
    for (int i = i0; i < 64 * 32;  i += stride)  dW2bf[i] = f2bf(dW2[i]);
    for (int i = i0; i < 128 * 64; i += stride)  dW3bf[i] = f2bf(dW3[i]);
    for (int i = i0; i < 2048 * 128; i += stride) dW4bf[i] = f2bf(dW4[i]);
}

// ---------------------------------------------------------------------------
// Encoder layer 1: dual GEMM over K=2048 (x and x_dot share W1 B-fragments),
// plus rank-1 treatment term, bias, sigmoid and sigmoid-derivative gate.
// Weight chunk streams in via async LDS DMA while the VALU converts x/x_dot
// f32 -> bf16 into LDS. Block: 64 rows x 128 cols; 8 waves.
// ---------------------------------------------------------------------------
__global__ void __launch_bounds__(256)
enc1_kernel(const float* __restrict__ x, const float* __restrict__ xd,
            const float* __restrict__ treat, const float* __restrict__ eW1,
            const float* __restrict__ eb1, const unsigned short* __restrict__ W1bf,
            unsigned short* __restrict__ a1bf, unsigned short* __restrict__ g1bf) {
    __shared__ unsigned short sX[64 * 32];
    __shared__ unsigned short sXd[64 * 32];
    __shared__ unsigned short sW[128 * 32];

    const int tid = threadIdx.x;
    const int rowBlock = blockIdx.x * 64;
    const int wid = tid >> 5, lane = tid & 31;
    const int sel = lane >> 4, ml = lane & 15;
    const int mBase = (wid >> 1) * 16;
    const int cg = wid & 1;                       // 2 col-groups x 64 cols

    const v8f vzero = {0.f, 0.f, 0.f, 0.f, 0.f, 0.f, 0.f, 0.f};
    v8f accA[4], accG[4];
#pragma unroll
    for (int t = 0; t < 4; ++t) { accA[t] = vzero; accG[t] = vzero; }

    for (int kk = 0; kk < DIN; kk += 32) {
        // W chunk: 128 rows x 32 cols, 64 contiguous bytes per row -> async DMA.
#if HAVE_ASYNC_LDS
        for (int i = tid; i < 512; i += 256) {     // 512 x 16B units
            int n = i >> 2, sub = (i & 3) * 8;
            async_copy_b128(W1bf + (size_t)n * DIN + kk + sub, sW + n * 32 + sub);
        }
#else
        for (int i = tid; i < 128 * 32; i += 256) {
            int n = i >> 5, c = i & 31;
            sW[i] = W1bf[(size_t)n * DIN + kk + c];
        }
#endif
        // x / x_dot chunk: f32 -> bf16 conversion on the VALU (overlaps W DMA).
        for (int i = tid; i < 64 * 32; i += 256) {
            int r = i >> 5, c = i & 31;
            size_t gi = (size_t)(rowBlock + r) * DIN + kk + c;
            sX[i]  = f2bf(x[gi]);
            sXd[i] = f2bf(xd[gi]);
        }
        tile_copy_fence();

        Frag fa, fg;
        load_a_frag(fa, sX,  mBase + ml, 32, 0, lane);
        load_a_frag(fg, sXd, mBase + ml, 32, 0, lane);
#pragma unroll
        for (int t = 0; t < 4; ++t) {
            Frag fb;
            load_b_frag(fb, sW, cg * 64 + t * 16 + ml, 32, 0, lane);
            accA[t] = wmma_bf16(fa, fb, accA[t]);
            accG[t] = wmma_bf16(fg, fb, accG[t]);
        }
        __syncthreads();
    }

    float trv[8];
#pragma unroll
    for (int v = 0; v < 8; ++v)
        trv[v] = treat[rowBlock + mBase + sel * 8 + v];

#pragma unroll
    for (int t = 0; t < 4; ++t) {
        int col = cg * 64 + t * 16 + ml;
        float wl = eW1[(size_t)col * 2049 + 2048];   // treatment column of eW1
        float b  = eb1[col];
#pragma unroll
        for (int v = 0; v < 8; ++v) {
            int row = rowBlock + mBase + sel * 8 + v;
            float preA = accA[t][v] + trv[v] * wl + b;
            float a = sigmoidf(preA);
            float preG = accG[t][v] + trv[v] * wl;
            float d = a * (1.0f - a) * preG;
            a1bf[(size_t)row * 128 + col] = f2bf(a);
            g1bf[(size_t)row * 128 + col] = f2bf(d);
        }
    }
}

// ---------------------------------------------------------------------------
// Generic dual GEMM layer: O = sigmoid(A@W.T + b), D = O*(1-O)*(G@W.T).
// Block: 64 rows x N cols; whole K resident in LDS (K <= 128); all three
// bf16 tiles arrive via async LDS DMA.
// ---------------------------------------------------------------------------
template<int K, int N>
__global__ void __launch_bounds__(256)
dual_layer_kernel(const unsigned short* __restrict__ Abf, const unsigned short* __restrict__ Gbf,
                  const unsigned short* __restrict__ Wbf, const float* __restrict__ bias,
                  unsigned short* __restrict__ Obf, unsigned short* __restrict__ Dbf) {
    __shared__ unsigned short sA[64 * K];
    __shared__ unsigned short sG[64 * K];
    __shared__ unsigned short sW[N * K];

    const int tid = threadIdx.x;
    const int rowBlock = blockIdx.x * 64;

    tile_copy(Abf + (size_t)rowBlock * K, sA, 64 * K / 8, tid);
    tile_copy(Gbf + (size_t)rowBlock * K, sG, 64 * K / 8, tid);
    tile_copy(Wbf, sW, N * K / 8, tid);
    tile_copy_fence();

    constexpr int NT = N / 32;                     // col tiles per wave
    const int wid = tid >> 5, lane = tid & 31;
    const int sel = lane >> 4, ml = lane & 15;
    const int mBase = (wid >> 1) * 16;
    const int cg = wid & 1;

    const v8f vzero = {0.f, 0.f, 0.f, 0.f, 0.f, 0.f, 0.f, 0.f};
    v8f accA[NT], accG[NT];
#pragma unroll
    for (int t = 0; t < NT; ++t) { accA[t] = vzero; accG[t] = vzero; }

    for (int kk = 0; kk < K; kk += 32) {
        Frag fa, fg;
        load_a_frag(fa, sA, mBase + ml, K, kk, lane);
        load_a_frag(fg, sG, mBase + ml, K, kk, lane);
#pragma unroll
        for (int t = 0; t < NT; ++t) {
            Frag fb;
            load_b_frag(fb, sW, cg * (N / 2) + t * 16 + ml, K, kk, lane);
            accA[t] = wmma_bf16(fa, fb, accA[t]);
            accG[t] = wmma_bf16(fg, fb, accG[t]);
        }
    }

#pragma unroll
    for (int t = 0; t < NT; ++t) {
        int col = cg * (N / 2) + t * 16 + ml;
        float b = bias[col];
#pragma unroll
        for (int v = 0; v < 8; ++v) {
            int row = rowBlock + mBase + sel * 8 + v;
            float o = sigmoidf(accA[t][v] + b);
            float d = o * (1.0f - o) * accG[t][v];
            Obf[(size_t)row * N + col] = f2bf(o);
            Dbf[(size_t)row * N + col] = f2bf(d);
        }
    }
}

// ---------------------------------------------------------------------------
// Encoder layer 4 (K=32,N=3) + SINDy library + decoder layer 1 (K=3,N=32) +
// loss_po / loss_tr / sindy_z partial sums. One thread per batch row.
// ---------------------------------------------------------------------------
__global__ void __launch_bounds__(256)
enc4_sindy_kernel(const unsigned short* __restrict__ a3bf, const unsigned short* __restrict__ g3bf,
                  const float* __restrict__ eW4, const float* __restrict__ eb4,
                  const float* __restrict__ dW1, const float* __restrict__ db1,
                  const float* __restrict__ coeff, const float* __restrict__ treat,
                  const float* __restrict__ sizev,
                  unsigned short* __restrict__ h1bf, unsigned short* __restrict__ gd1bf,
                  float* __restrict__ z_out, float* accum) {
    __shared__ float red[256];
    const int tid = threadIdx.x;
    const int row = blockIdx.x * 256 + tid;

    float a3[32], g3[32];
#pragma unroll
    for (int k = 0; k < 32; ++k) {
        a3[k] = bf2f(a3bf[(size_t)row * 32 + k]);
        g3[k] = bf2f(g3bf[(size_t)row * 32 + k]);
    }

    float z[3], zdt[3];
#pragma unroll
    for (int j = 0; j < 3; ++j) {
        float s1 = 0.f, s2 = 0.f;
#pragma unroll
        for (int k = 0; k < 32; ++k) {
            float w = eW4[j * 32 + k];
            s1 += a3[k] * w;
            s2 += g3[k] * w;
        }
        z[j]  = s1 + eb4[j];
        zdt[j] = s2;                               // z_dot_true
        z_out[(size_t)row * 3 + j] = z[j];
    }

    const float s = z[0], d = z[1], t = z[2];
    float theta[7] = {1.0f, s, s * s, s * d, s * t, s * s * d, s * s * t};
    float zdp[3];
#pragma unroll
    for (int j = 0; j < 3; ++j) {
        float acc = 0.f;
#pragma unroll
        for (int i = 0; i < 7; ++i) acc += theta[i] * coeff[i * 3 + j];
        zdp[j] = acc;                              // z_dot_pred
    }

    // decoder layer 1 forward + chain derivative
#pragma unroll
    for (int n = 0; n < 32; ++n) {
        float pre = db1[n], g = 0.f;
#pragma unroll
        for (int j = 0; j < 3; ++j) {
            float w = dW1[n * 3 + j];
            pre += z[j] * w;
            g   += zdp[j] * w;
        }
        float h = sigmoidf(pre);
        h1bf[(size_t)row * 32 + n]  = f2bf(h);
        gd1bf[(size_t)row * 32 + n] = f2bf(h * (1.0f - h) * g);
    }

    float szv = sizev[row], trv = treat[row];
    float lpo = (z[0] - szv) * (z[0] - szv);
    float lt  = fmaxf(t, 0.0f) + log1pf(expf(-fabsf(t))) - t * trv;  // logaddexp(0,t)-t*tr
    float lsz = 0.f;
#pragma unroll
    for (int j = 0; j < 3; ++j) {
        float e = zdt[j] - zdp[j];
        lsz += e * e;
    }
    block_reduce_atomic(lpo, red, tid, &accum[0]);
    block_reduce_atomic(lt,  red, tid, &accum[1]);
    block_reduce_atomic(lsz, red, tid, &accum[4]);
}

// ---------------------------------------------------------------------------
// Decoder layer 4 (K=128, N=2048): x_hat = h3@dW4.T + db4, and fused
// recon / sindy_x reductions (x_dot_pred never hits HBM).
// Block tile: 64 rows x 64 cols; tiles via async LDS DMA.
// ---------------------------------------------------------------------------
__global__ void __launch_bounds__(256)
dec4_kernel(const unsigned short* __restrict__ h3bf, const unsigned short* __restrict__ gd3bf,
            const unsigned short* __restrict__ W4bf, const float* __restrict__ db4,
            const float* __restrict__ x, const float* __restrict__ xd,
            float* __restrict__ xhat, float* accum) {
    __shared__ unsigned short sA[64 * 128];
    __shared__ unsigned short sG[64 * 128];
    __shared__ unsigned short sW[64 * 128];
    __shared__ float red[256];

    const int tid = threadIdx.x;
    const int rowBlock = blockIdx.x * 64;
    const int colBlock = blockIdx.y * 64;

    tile_copy(h3bf  + (size_t)rowBlock * 128, sA, 1024, tid);
    tile_copy(gd3bf + (size_t)rowBlock * 128, sG, 1024, tid);
    tile_copy(W4bf  + (size_t)colBlock * 128, sW, 1024, tid);
    tile_copy_fence();

    const int wid = tid >> 5, lane = tid & 31;
    const int sel = lane >> 4, ml = lane & 15;
    const int mBase = (wid >> 1) * 16;
    const int cg = wid & 1;

    const v8f vzero = {0.f, 0.f, 0.f, 0.f, 0.f, 0.f, 0.f, 0.f};
    v8f accA[2] = {vzero, vzero};
    v8f accG[2] = {vzero, vzero};

    for (int kk = 0; kk < 128; kk += 32) {
        Frag fa, fg;
        load_a_frag(fa, sA, mBase + ml, 128, kk, lane);
        load_a_frag(fg, sG, mBase + ml, 128, kk, lane);
#pragma unroll
        for (int t = 0; t < 2; ++t) {
            Frag fb;
            load_b_frag(fb, sW, cg * 32 + t * 16 + ml, 128, kk, lane);
            accA[t] = wmma_bf16(fa, fb, accA[t]);
            accG[t] = wmma_bf16(fg, fb, accG[t]);
        }
    }

    float lr = 0.f, lsx = 0.f;
#pragma unroll
    for (int t = 0; t < 2; ++t) {
        int col = colBlock + cg * 32 + t * 16 + ml;
        float b = db4[col];
#pragma unroll
        for (int v = 0; v < 8; ++v) {
            int row = rowBlock + mBase + sel * 8 + v;
            size_t idx = (size_t)row * DIN + col;
            float xh = accA[t][v] + b;
            xhat[idx] = xh;
            float e1 = x[idx] - xh;
            lr += e1 * e1;
            float e2 = xd[idx] - accG[t][v];       // x_dot_pred, no bias
            lsx += e2 * e2;
        }
    }
    block_reduce_atomic(lr,  red, tid, &accum[2]);
    block_reduce_atomic(lsx, red, tid, &accum[3]);
}

__global__ void finalize_kernel(const float* accum, const float* coeff, float* sc) {
    if (threadIdx.x == 0 && blockIdx.x == 0) {
        float l1 = 0.f;
        for (int i = 0; i < 21; ++i) l1 += fabsf(coeff[i]);
        const float fB = 16384.0f;
        sc[0] = accum[0] / fB;                     // loss_po
        sc[1] = accum[1] / fB;                     // loss_tr
        sc[2] = accum[2] / (fB * 2048.0f);         // recon
        sc[3] = accum[3] / (fB * 2048.0f);         // sindy_x
        sc[4] = accum[4] / (fB * 3.0f);            // sindy_z
        sc[5] = l1 / 21.0f;                        // l1
    }
}

// ---------------------------------------------------------------------------
extern "C" void kernel_launch(void* const* d_in, const int* in_sizes, int n_in,
                              void* d_out, int out_size, void* d_ws, size_t ws_size,
                              hipStream_t stream) {
    const float* x      = (const float*)d_in[0];
    const float* x_dot  = (const float*)d_in[1];
    const float* treat  = (const float*)d_in[2];
    const float* sizev  = (const float*)d_in[3];
    const float* eW1 = (const float*)d_in[4],  *eb1 = (const float*)d_in[5];
    const float* eW2 = (const float*)d_in[6],  *eb2 = (const float*)d_in[7];
    const float* eW3 = (const float*)d_in[8],  *eb3 = (const float*)d_in[9];
    const float* eW4 = (const float*)d_in[10], *eb4 = (const float*)d_in[11];
    const float* dW1 = (const float*)d_in[12], *db1 = (const float*)d_in[13];
    const float* dW2 = (const float*)d_in[14], *db2 = (const float*)d_in[15];
    const float* dW3 = (const float*)d_in[16], *db3 = (const float*)d_in[17];
    const float* dW4 = (const float*)d_in[18], *db4 = (const float*)d_in[19];
    const float* coeff = (const float*)d_in[20];

    const size_t B = BATCH;
    char* p = (char*)d_ws;
    auto alloc = [&](size_t bytes) -> void* {
        void* r = (void*)p;
        p += (bytes + 255) & ~(size_t)255;
        return r;
    };
    float* accum = (float*)alloc(64 * sizeof(float));
    unsigned short* a1bf  = (unsigned short*)alloc(B * 128 * 2);
    unsigned short* g1bf  = (unsigned short*)alloc(B * 128 * 2);
    unsigned short* a2bf  = (unsigned short*)alloc(B * 64 * 2);
    unsigned short* g2bf  = (unsigned short*)alloc(B * 64 * 2);
    unsigned short* a3bf  = (unsigned short*)alloc(B * 32 * 2);
    unsigned short* g3bf  = (unsigned short*)alloc(B * 32 * 2);
    unsigned short* h1bf  = (unsigned short*)alloc(B * 32 * 2);
    unsigned short* gd1bf = (unsigned short*)alloc(B * 32 * 2);
    unsigned short* h2bf  = (unsigned short*)alloc(B * 64 * 2);
    unsigned short* gd2bf = (unsigned short*)alloc(B * 64 * 2);
    unsigned short* h3bf  = (unsigned short*)alloc(B * 128 * 2);
    unsigned short* gd3bf = (unsigned short*)alloc(B * 128 * 2);
    unsigned short* W1bf  = (unsigned short*)alloc((size_t)128 * 2048 * 2);
    unsigned short* W2bf  = (unsigned short*)alloc((size_t)64 * 128 * 2);
    unsigned short* W3bf  = (unsigned short*)alloc((size_t)32 * 64 * 2);
    unsigned short* dW2bf = (unsigned short*)alloc((size_t)64 * 32 * 2);
    unsigned short* dW3bf = (unsigned short*)alloc((size_t)128 * 64 * 2);
    unsigned short* dW4bf = (unsigned short*)alloc((size_t)2048 * 128 * 2);

    float* out   = (float*)d_out;
    float* z_out = out;                          // B*3
    float* xhat  = out + B * 3;                  // B*2048
    float* sc    = out + B * 3 + B * (size_t)DIN; // 6 scalars

    pack_kernel<<<512, 256, 0, stream>>>(eW1, eW2, eW3, dW2, dW3, dW4,
                                         W1bf, W2bf, W3bf, dW2bf, dW3bf, dW4bf, accum);
    enc1_kernel<<<BATCH / 64, 256, 0, stream>>>(x, x_dot, treat, eW1, eb1, W1bf, a1bf, g1bf);
    dual_layer_kernel<128, 64><<<BATCH / 64, 256, 0, stream>>>(a1bf, g1bf, W2bf, eb2, a2bf, g2bf);
    dual_layer_kernel<64, 32><<<BATCH / 64, 256, 0, stream>>>(a2bf, g2bf, W3bf, eb3, a3bf, g3bf);
    enc4_sindy_kernel<<<BATCH / 256, 256, 0, stream>>>(a3bf, g3bf, eW4, eb4, dW1, db1, coeff,
                                                       treat, sizev, h1bf, gd1bf, z_out, accum);
    dual_layer_kernel<32, 64><<<BATCH / 64, 256, 0, stream>>>(h1bf, gd1bf, dW2bf, db2, h2bf, gd2bf);
    dual_layer_kernel<64, 128><<<BATCH / 64, 256, 0, stream>>>(h2bf, gd2bf, dW3bf, db3, h3bf, gd3bf);
    dec4_kernel<<<dim3(BATCH / 64, DIN / 64), 256, 0, stream>>>(h3bf, gd3bf, dW4bf, db4,
                                                                x, x_dot, xhat, accum);
    finalize_kernel<<<1, 32, 0, stream>>>(accum, coeff, sc);
}
